// SPFBlock_54528904790280
// MI455X (gfx1250) — matrix operand order repaired and checked by reading, not compiled, not run on hardware
//
#include <hip/hip_runtime.h>
#include <hip/hip_bf16.h>
#include <math.h>

typedef __attribute__((ext_vector_type(16))) _Float16 v16h;
typedef __attribute__((ext_vector_type(8)))  _Float16 v8h;
typedef __attribute__((ext_vector_type(2)))  _Float16 v2h;
typedef __attribute__((ext_vector_type(8)))  float    v8f;

#define L_TOT   16384
#define C_IN    128
#define NH      4
#define NPH     32
#define MTRI    528
#define CPOOL   2112
#define DIMO    64
#define KB_DR   66   // 2112 / 32
#define KB_FC   2    // 64 / 32

// ---------------------------------------------------------------------------
// Kernel 1: LayerNorm over 128 channels (NCHW, per spatial position) + channel
// shuffle.  Output t[b][l][newc], newc = (c%4)*32 + c/4.
// Block: 128 threads = 128 channels; each block handles 32 consecutive l.
// ---------------------------------------------------------------------------
__global__ __launch_bounds__(128)
void spf_ln_shuffle(const float* __restrict__ x,
                    const float* __restrict__ g1,
                    const float* __restrict__ b1,
                    float* __restrict__ t_ws)
{
    __shared__ float xl[32 * 132];   // [p][c], pad 132 to dodge bank conflicts
    __shared__ float tl[32 * 132];
    __shared__ float gs[128], bs[128];

    const int tid = threadIdx.x;
    const int b   = blockIdx.x >> 9;           // 512 chunks per batch
    const int l0  = (blockIdx.x & 511) << 5;   // 32 positions per block

    gs[tid] = g1[tid];
    bs[tid] = b1[tid];

    // channel tid: load 32 consecutive positions (contiguous 128B)
    const float* xp = x + ((size_t)(b * C_IN + tid)) * L_TOT + l0;
#pragma unroll
    for (int k = 0; k < 8; ++k) {
        float4 q = ((const float4*)xp)[k];
        int p = 4 * k;
        xl[(p + 0) * 132 + tid] = q.x;
        xl[(p + 1) * 132 + tid] = q.y;
        xl[(p + 2) * 132 + tid] = q.z;
        xl[(p + 3) * 132 + tid] = q.w;
    }
    __syncthreads();

    // reduction: 4 threads per position, 32 channels each
    const int p  = tid >> 2;
    const int qt = tid & 3;
    const float* row = &xl[p * 132 + qt * 32];
    float s = 0.f, ss = 0.f;
#pragma unroll 8
    for (int c = 0; c < 32; ++c) { float v = row[c]; s += v; ss += v * v; }
    s  += __shfl_xor(s, 1);  s  += __shfl_xor(s, 2);
    ss += __shfl_xor(ss, 1); ss += __shfl_xor(ss, 2);
    const float mu   = s * (1.f / 128.f);
    const float var  = ss * (1.f / 128.f) - mu * mu;
    const float rstd = rsqrtf(var + 1e-5f);

    // normalize + shuffle into tl[p][newc]
#pragma unroll 8
    for (int c0 = 0; c0 < 32; ++c0) {
        int c  = qt * 32 + c0;
        int nc = (c & 3) * 32 + (c >> 2);
        tl[p * 132 + nc] = (xl[p * 132 + c] - mu) * rstd * gs[c] + bs[c];
    }
    __syncthreads();

    // coalesced write-out
    for (int idx = tid; idx < 32 * 128; idx += 128) {
        int pp = idx >> 7, cc = idx & 127;
        t_ws[((size_t)(b * L_TOT + l0 + pp)) * C_IN + cc] = tl[pp * 132 + cc];
    }
}

// ---------------------------------------------------------------------------
// Kernel 2: pack dr_w (64x2112) and fc1_w (64x64) into f16 in WMMA A-fragment
// order.  ISA 16-bit A 16x32 layout: lane = half*16 + m; element e ->
// K = (e<8 ? e : e+8) + 8*half.  Stored as [w][kb][lane][16] so each lane
// loads one contiguous v16h per K step.
// ---------------------------------------------------------------------------
__global__ __launch_bounds__(256)
void spf_pack(const float* __restrict__ dr_w,
              const float* __restrict__ fc1_w,
              _Float16* __restrict__ a_packed,
              _Float16* __restrict__ f_packed)
{
    const int idx = blockIdx.x * 256 + threadIdx.x;
    const int NA  = DIMO * CPOOL;              // 135168
    if (idx < NA) {
        int e = idx & 15, lane = (idx >> 4) & 31, t = idx >> 9;
        int kb = t % KB_DR, w = t / KB_DR;
        int m = lane & 15, hf = lane >> 4;
        int k = kb * 32 + (e < 8 ? e : e + 8) + 8 * hf;
        a_packed[idx] = (_Float16)dr_w[(w * 16 + m) * CPOOL + k];
    } else {
        int j = idx - NA;
        if (j < DIMO * DIMO) {
            int e = j & 15, lane = (j >> 4) & 31, t = j >> 9;
            int kb = t & 1, w = t >> 1;
            int m = lane & 15, hf = lane >> 4;
            int k = kb * 32 + (e < 8 ? e : e + 8) + 8 * hf;
            f_packed[j] = (_Float16)fc1_w[(w * 16 + m) * DIMO + k];
        }
    }
}

// ---------------------------------------------------------------------------
// Kernel 3: closed-form attention logits.
//   ||pf||^2 = (S2^2 + S4)/2 ,  <pf_l,pf_mid> = ((t.u)^2 + sum t^2 u^2)/2
// One thread per (b,h,l).
// ---------------------------------------------------------------------------
__global__ __launch_bounds__(256)
void spf_logits(const float* __restrict__ t_ws, float* __restrict__ logits)
{
    const int l  = blockIdx.x * 256 + threadIdx.x;
    const int bh = blockIdx.y;
    const int b  = bh >> 2, h = bh & 3;

    const float4* tp = (const float4*)(t_ws + ((size_t)(b * L_TOT + l)) * C_IN + h * NPH);
    const float4* up = (const float4*)(t_ws + ((size_t)(b * L_TOT + (L_TOT / 2))) * C_IN + h * NPH);

    float ta[32], ua[32];
#pragma unroll
    for (int k = 0; k < 8; ++k) { ((float4*)ta)[k] = tp[k]; ((float4*)ua)[k] = up[k]; }

    float d = 0.f, q = 0.f, s2 = 0.f, s4 = 0.f, s2u = 0.f, s4u = 0.f;
#pragma unroll 8
    for (int i = 0; i < 32; ++i) {
        float t = ta[i], u = ua[i];
        float t2 = t * t, u2 = u * u;
        d += t * u; q += t2 * u2;
        s2 += t2; s4 += t2 * t2;
        s2u += u2; s4u += u2 * u2;
    }
    float nl  = 0.5f * (s2 * s2 + s4);
    float nm  = 0.5f * (s2u * s2u + s4u);
    float dot = 0.5f * (d * d + q);
    logits[bh * L_TOT + l] = dot * rsqrtf(nl * nm);
}

// ---------------------------------------------------------------------------
// Kernel 4: softmax over L per (b,h); write s = softmax * L.
// ---------------------------------------------------------------------------
__global__ __launch_bounds__(256)
void spf_softmax(const float* __restrict__ logits, float* __restrict__ s_ws)
{
    __shared__ float red[8];
    __shared__ float bval;
    const int bh  = blockIdx.x;
    const int tid = threadIdx.x;
    const float* lp = logits + bh * L_TOT;

    float mx = -1e30f;
    for (int i = tid; i < L_TOT; i += 256) mx = fmaxf(mx, lp[i]);
    for (int off = 16; off; off >>= 1) mx = fmaxf(mx, __shfl_xor(mx, off));
    if ((tid & 31) == 0) red[tid >> 5] = mx;
    __syncthreads();
    if (tid == 0) { float m = red[0]; for (int i = 1; i < 8; ++i) m = fmaxf(m, red[i]); bval = m; }
    __syncthreads();
    mx = bval;

    float sum = 0.f;
    for (int i = tid; i < L_TOT; i += 256) sum += expf(lp[i] - mx);
    for (int off = 16; off; off >>= 1) sum += __shfl_xor(sum, off);
    if ((tid & 31) == 0) red[tid >> 5] = sum;
    __syncthreads();
    if (tid == 0) { float s = 0.f; for (int i = 0; i < 8; ++i) s += red[i]; bval = (float)L_TOT / s; }
    __syncthreads();
    const float scl = bval;

    float* sp = s_ws + bh * L_TOT;
    for (int i = tid; i < L_TOT; i += 256) sp[i] = expf(lp[i] - mx) * scl;
}

// ---------------------------------------------------------------------------
// Kernel 5: fused pf-generation + dr GEMM (WMMA) + dr_b + norm2 + fc1 GEMM
// (WMMA) + GELU.
// One workgroup = 32 positions x all 64 outputs; 4 waves, wave w owns output
// rows [w*16, w*16+16) and BOTH 16-position N tiles (one A frag feeds two
// WMMAs -> half the A-operand L2 traffic + 2x ILP on the matrix pipe).
// pf is built in 2 K-chunks (heads {0,1} then {2,3}) to keep LDS ~100KB.
// ---------------------------------------------------------------------------
__global__ __launch_bounds__(128)
void spf_gemm(const float* __restrict__ t_ws,
              const float* __restrict__ s_ws,
              const _Float16* __restrict__ a_packed,
              const _Float16* __restrict__ f_packed,
              const float* __restrict__ dr_b,
              const float* __restrict__ n2w,
              const float* __restrict__ n2b,
              const float* __restrict__ f1b,
              float* __restrict__ out)
{
    // pf chunk: [pos 0..31][c' 0..1055], row stride 1064 halfs (2128B = 133*16B)
    __shared__ __attribute__((aligned(16))) _Float16 pf[32 * 1064];   // 68096 B
    __shared__ float t_tile[32 * 133];                                // 17024 B
    __shared__ float s_tile[NH * 32];                                 // [h][p]
    __shared__ float z_lds[32 * 69];                                  // [pos][o]
    __shared__ __attribute__((aligned(16))) _Float16 zn_lds[32 * 72]; // [pos][o]
    __shared__ float drb_s[64], n2w_s[64], n2b_s[64], f1b_s[64];

    const int tid = threadIdx.x;
    const int b   = blockIdx.x >> 9;           // 512 tiles of 32 per batch
    const int l0  = (blockIdx.x & 511) << 5;

    // ---- stage t tile (32x128), scales, biases ----
    for (int idx = tid; idx < 32 * C_IN; idx += 128) {
        int p = idx >> 7, ch = idx & 127;
        t_tile[p * 133 + ch] = t_ws[((size_t)(b * L_TOT + l0 + p)) * C_IN + ch];
    }
    s_tile[tid] = s_ws[((b * NH + (tid >> 5)) * L_TOT) + l0 + (tid & 31)];
    if (tid < 64) {
        drb_s[tid] = dr_b[tid]; n2w_s[tid] = n2w[tid];
        n2b_s[tid] = n2b[tid];  f1b_s[tid] = f1b[tid];
    }

    const int lane   = tid & 31;
    const int w      = tid >> 5;      // wave id -> M block (also pf-gen sub id)
    const int col    = lane & 15;     // N within tile for B/C fragments
    const int halfid = lane >> 4;

    // pf-gen thread role: position p = lane, sub-block = wave id
    const int gp  = lane;             // position 0..31
    const int hl  = w >> 1;           // head-within-chunk 0/1
    const int par = w & 1;            // first/second 264-pair half of the head

    v8f acc0 = {0.f, 0.f, 0.f, 0.f, 0.f, 0.f, 0.f, 0.f};
    v8f acc1 = {0.f, 0.f, 0.f, 0.f, 0.f, 0.f, 0.f, 0.f};
    const v16h* apb = (const v16h*)a_packed + ((size_t)w * KB_DR * 32 + lane);

    for (int g = 0; g < 2; ++g) {
        __syncthreads();   // staging done (g=0) / previous chunk consumed (g=1)

        // ---- generate pf chunk: heads 2g and 2g+1 ----
        {
            const int h = 2 * g + hl;
            const float* tpb = &t_tile[gp * 133 + h * NPH];
            const float sc = s_tile[h * 32 + gp];
            // m=0 -> (i,j)=(0,0);  m=264 -> (9,21)   [row starts: 32i - i(i-1)/2]
            int i = par ? 9 : 0;
            int j = par ? 21 : 0;
            float ti = tpb[i] * sc;
            _Float16* prow = &pf[gp * 1064 + hl * MTRI];
            const int m0 = par * 264;
#pragma unroll 4
            for (int m = m0; m < m0 + 264; m += 2) {
                float v0 = ti * tpb[j];
                ++j; if (j == NPH) { ++i; j = i; ti = tpb[i] * sc; }
                float v1 = ti * tpb[j];
                ++j; if (j == NPH) { ++i; j = i; ti = tpb[i] * sc; }
                v2h pk; pk[0] = (_Float16)v0; pk[1] = (_Float16)v1;
                *(v2h*)&prow[m] = pk;
            }
        }
        __syncthreads();

        // ---- WMMA over this K chunk: 1 A frag feeds 2 N tiles ----
        for (int kbl = 0; kbl < 33; ++kbl) {
            v16h afrag = apb[(g * 33 + kbl) * 32];
            const _Float16* bp0 = &pf[col * 1064 + kbl * 32 + halfid * 8];
            const _Float16* bp1 = &pf[(16 + col) * 1064 + kbl * 32 + halfid * 8];
            v8h b0lo = *(const v8h*)bp0;
            v8h b0hi = *(const v8h*)(bp0 + 16);
            v8h b1lo = *(const v8h*)bp1;
            v8h b1hi = *(const v8h*)(bp1 + 16);
            v16h bf0 = __builtin_shufflevector(b0lo, b0hi,
                        0, 1, 2, 3, 4, 5, 6, 7, 8, 9, 10, 11, 12, 13, 14, 15);
            v16h bf1 = __builtin_shufflevector(b1lo, b1hi,
                        0, 1, 2, 3, 4, 5, 6, 7, 8, 9, 10, 11, 12, 13, 14, 15);
            acc0 = __builtin_amdgcn_wmma_f32_16x16x32_f16(
                       false, afrag, false, bf0, (short)0, acc0, false, false);
            acc1 = __builtin_amdgcn_wmma_f32_16x16x32_f16(
                       false, afrag, false, bf1, (short)0, acc1, false, false);
        }
    }

    // ---- z -> LDS with dr bias (both N tiles) ----
#pragma unroll
    for (int r = 0; r < 8; ++r) {
        int o = w * 16 + halfid * 8 + r;
        z_lds[col * 69 + o]        = acc0[r] + drb_s[o];
        z_lds[(16 + col) * 69 + o] = acc1[r] + drb_s[o];
    }
    __syncthreads();

    // ---- norm2 over 64 channels per position, emit f16 B operand ----
    if (tid < 32) {
        const float* zr = &z_lds[tid * 69];
        float mu = 0.f;
        for (int o = 0; o < DIMO; ++o) mu += zr[o];
        mu *= (1.f / 64.f);
        float var = 0.f;
        for (int o = 0; o < DIMO; ++o) { float d = zr[o] - mu; var += d * d; }
        var *= (1.f / 64.f);
        float rstd = rsqrtf(var + 1e-5f);
        _Float16* zo = &zn_lds[tid * 72];
        for (int o = 0; o < DIMO; ++o)
            zo[o] = (_Float16)((zr[o] - mu) * rstd * n2w_s[o] + n2b_s[o]);
    }
    __syncthreads();

    // ---- GEMM 2: fc1 (64x64), 2 K steps x 2 N tiles ----
    v8f f0 = {0.f, 0.f, 0.f, 0.f, 0.f, 0.f, 0.f, 0.f};
    v8f f1 = {0.f, 0.f, 0.f, 0.f, 0.f, 0.f, 0.f, 0.f};
    const v16h* fpb = (const v16h*)f_packed + ((size_t)w * KB_FC * 32 + lane);
#pragma unroll
    for (int kb = 0; kb < KB_FC; ++kb) {
        v16h afrag = fpb[kb * 32];
        const _Float16* bp0 = &zn_lds[col * 72 + kb * 32 + halfid * 8];
        const _Float16* bp1 = &zn_lds[(16 + col) * 72 + kb * 32 + halfid * 8];
        v8h b0lo = *(const v8h*)bp0;
        v8h b0hi = *(const v8h*)(bp0 + 16);
        v8h b1lo = *(const v8h*)bp1;
        v8h b1hi = *(const v8h*)(bp1 + 16);
        v16h bf0 = __builtin_shufflevector(b0lo, b0hi,
                    0, 1, 2, 3, 4, 5, 6, 7, 8, 9, 10, 11, 12, 13, 14, 15);
        v16h bf1 = __builtin_shufflevector(b1lo, b1hi,
                    0, 1, 2, 3, 4, 5, 6, 7, 8, 9, 10, 11, 12, 13, 14, 15);
        f0 = __builtin_amdgcn_wmma_f32_16x16x32_f16(
                 false, afrag, false, bf0, (short)0, f0, false, false);
        f1 = __builtin_amdgcn_wmma_f32_16x16x32_f16(
                 false, afrag, false, bf1, (short)0, f1, false, false);
    }

    // ---- exact GELU + store NCHW ----
#pragma unroll
    for (int r = 0; r < 8; ++r) {
        int o = w * 16 + halfid * 8 + r;
        size_t base = ((size_t)(b * DIMO + o)) * L_TOT + l0;
        float x0 = f0[r] + f1b_s[o];
        float x1 = f1[r] + f1b_s[o];
        out[base + col]      = 0.5f * x0 * (1.0f + erff(x0 * 0.70710678118654752f));
        out[base + 16 + col] = 0.5f * x1 * (1.0f + erff(x1 * 0.70710678118654752f));
    }
}

// ---------------------------------------------------------------------------
extern "C" void kernel_launch(void* const* d_in, const int* in_sizes, int n_in,
                              void* d_out, int out_size, void* d_ws, size_t ws_size,
                              hipStream_t stream)
{
    (void)in_sizes; (void)n_in; (void)out_size; (void)ws_size;

    const float* x       = (const float*)d_in[0];
    const float* norm1_w = (const float*)d_in[1];
    const float* norm1_b = (const float*)d_in[2];
    const float* dr_w    = (const float*)d_in[3];
    const float* dr_b    = (const float*)d_in[4];
    const float* norm2_w = (const float*)d_in[5];
    const float* norm2_b = (const float*)d_in[6];
    const float* fc1_w   = (const float*)d_in[7];
    const float* fc1_b   = (const float*)d_in[8];
    float* out = (float*)d_out;

    char* ws = (char*)d_ws;
    float*    t_ws     = (float*)(ws);                       // 4*16384*128*4  = 33,554,432 B
    float*    logits   = (float*)(ws + 33554432);            // 16*16384*4    =  1,048,576 B
    float*    s_ws     = (float*)(ws + 34603008);            //                  1,048,576 B
    _Float16* a_packed = (_Float16*)(ws + 35651584);         // 64*2112*2     =    270,336 B
    _Float16* f_packed = (_Float16*)(ws + 35921920);         // 64*64*2       =      8,192 B

    // 1) layernorm + shuffle -> t
    spf_ln_shuffle<<<dim3(4 * (L_TOT / 32)), dim3(128), 0, stream>>>(x, norm1_w, norm1_b, t_ws);

    // 2) weight packing to f16 A-fragment layout
    {
        int total = DIMO * CPOOL + DIMO * DIMO;
        spf_pack<<<dim3((total + 255) / 256), dim3(256), 0, stream>>>(dr_w, fc1_w, a_packed, f_packed);
    }

    // 3) closed-form CTM logits
    spf_logits<<<dim3(L_TOT / 256, 16), dim3(256), 0, stream>>>(t_ws, logits);

    // 4) softmax -> scales
    spf_softmax<<<dim3(16), dim3(256), 0, stream>>>(logits, s_ws);

    // 5) fused pf-gen + WMMA GEMMs + norm2 + GELU
    spf_gemm<<<dim3(4 * (L_TOT / 32)), dim3(128), 0, stream>>>(
        t_ws, s_ws, a_packed, f_packed, dr_b, norm2_w, norm2_b, fc1_b, out);
}
